// MultiHeadSelfAttentionFused_54915451846833
// MI455X (gfx1250) — compile-verified, matrix operand order checked by reading, not run on hardware
//
#include <hip/hip_runtime.h>
#include <hip/hip_bf16.h>

// ---------------------------------------------------------------------------
// Fused attention (reference semantics) on gfx1250, bf16 WMMA everywhere.
//   q = X_q Wq^T + bq ; k = X_k Wk^T + bk                          (NT gemm)
//   vT = (X_v Wv^T + bv)^T                                          (NT gemm, transposed writeback)
//   scores = q k^T * SCALE                                          (NT gemm, batched)
//   attn   = softmax(scores, axis=QUERY)                            (column softmax)
//   ctx    = attn @ V   == NT gemm against vT                       (NT gemm, batched)
//   out    = ctx Wo^T + bo                                          (NT gemm, fp32 out)
// All GEMMs share one NT kernel with async global->LDS double-buffered staging.
// ---------------------------------------------------------------------------

typedef __attribute__((ext_vector_type(16))) __bf16 v16bf;
typedef __attribute__((ext_vector_type(8)))  __bf16 v8bf;
typedef __attribute__((ext_vector_type(8)))  float  v8f;
typedef __attribute__((ext_vector_type(4)))  int    v4i;

typedef __attribute__((address_space(1))) v4i as1_v4i;
typedef __attribute__((address_space(3))) v4i as3_v4i;

#define BM 128          // block tile M (4 waves x 32 rows)
#define BN 64           // block tile N
#define BK 32           // K step (one bf16 WMMA K depth)
#define LDSS 40         // bf16 elems per LDS row: 32 data + 8 pad (80 B, 16B aligned)

#if __has_builtin(__builtin_amdgcn_global_load_async_to_lds_b128)
#define HAVE_ASYNC_BUILTIN 1
#else
#define HAVE_ASYNC_BUILTIN 0
#endif

__device__ __forceinline__ void async_copy_b128(void* lds_ptr, const void* gptr) {
#if HAVE_ASYNC_BUILTIN
  __builtin_amdgcn_global_load_async_to_lds_b128(
      (as1_v4i*)gptr, (as3_v4i*)lds_ptr, 0, 0);
#else
  asm volatile("global_load_async_to_lds_b128 %0, %1, off"
               :: "v"((unsigned)(unsigned long long)lds_ptr),
                  "v"((unsigned long long)gptr)
               : "memory");
#endif
}

__device__ __forceinline__ void wait_async0() {
#if __has_builtin(__builtin_amdgcn_s_wait_asynccnt)
  __builtin_amdgcn_s_wait_asynccnt(0);
#else
  asm volatile("s_wait_asynccnt 0x0" ::: "memory");
#endif
}

// ---------------------------------------------------------------------------
__global__ __launch_bounds__(128)
void cvt_f32_to_bf16_kernel(const float* __restrict__ in,
                            __bf16* __restrict__ out, long n) {
  long i = (long)blockIdx.x * blockDim.x + threadIdx.x;
  const long stride = (long)gridDim.x * blockDim.x;
  for (; i < n; i += stride) out[i] = (__bf16)in[i];
}

// ---------------------------------------------------------------------------
// NT GEMM: C[M,N] = alpha * A[M,K] x B[N,K]^T (+ bias[n])
//   A row-major, leading dim ldA; B row-major [N,K], leading dim ldB.
//   OUT_T: store transposed, C[col*ldC + row] (bf16 only use-case).
// Block: 128 threads = 4 waves; block tile 128x64; wave tile 32x64
//   (2 m-subtiles x 4 n-subtiles = 8 WMMA accumulators).
// Staging: async global->LDS b128, double-buffered, one barrier per K-step.
// ---------------------------------------------------------------------------
template<bool OUT_BF16, bool ADD_BIAS, bool OUT_T>
__global__ __launch_bounds__(128)
void gemm_nt_wmma_bf16_kernel(const __bf16* __restrict__ A, long ldA, long strideA,
                              const __bf16* __restrict__ Bm, long ldB, long strideB,
                              void* __restrict__ Cv, long ldC, long strideC,
                              const float* __restrict__ bias,
                              int K, float alpha) {
  __shared__ __bf16 ldsA[2][BM * LDSS];
  __shared__ __bf16 ldsB[2][BN * LDSS];

  const int bz = blockIdx.z;
  const __bf16* Ab = A  + (long)bz * strideA;
  const __bf16* Bb = Bm + (long)bz * strideB;

  const int tid  = threadIdx.x;   // 0..127
  const int wave = tid >> 5;
  const int lane = tid & 31;
  const int l16  = lane & 15;
  const int hi16 = lane >> 4;

  const long mBlock = (long)blockIdx.y * BM;
  const long nBlock = (long)blockIdx.x * BN;

  v8f acc[2][4] = {};

  // ---- async stage one K-tile (A: 128x32, B: 64x32 bf16) into buffer `buf`
  auto stage = [&](int buf, int k0) {
    // A tile: 512 16-byte chunks, 4 per thread
#pragma unroll
    for (int i = 0; i < 4; ++i) {
      const int c   = tid + 128 * i;
      const int row = c >> 2;             // 0..127
      const int cc  = (c & 3) * 8;        // bf16 offset in row: 0,8,16,24
      const __bf16* g = Ab + (mBlock + row) * ldA + k0 + cc;
      async_copy_b128(&ldsA[buf][row * LDSS + cc], g);
    }
    // B tile: 256 16-byte chunks, 2 per thread
#pragma unroll
    for (int i = 0; i < 2; ++i) {
      const int c   = tid + 128 * i;
      const int row = c >> 2;             // 0..63
      const int cc  = (c & 3) * 8;
      const __bf16* g = Bb + (nBlock + row) * ldB + k0 + cc;
      async_copy_b128(&ldsB[buf][row * LDSS + cc], g);
    }
  };

  // ---- consume one staged K-tile: 8 WMMAs per wave
  auto compute = [&](int buf) {
    const __bf16* lA = ldsA[buf];
    const __bf16* lB = ldsB[buf];
    v16bf af[2];
#pragma unroll
    for (int mt = 0; mt < 2; ++mt) {
      const int aRow = wave * 32 + mt * 16 + l16;
      v8bf lo = *(const v8bf*)&lA[aRow * LDSS + hi16 * 8];       // K 0-7 / 8-15
      v8bf hi = *(const v8bf*)&lA[aRow * LDSS + 16 + hi16 * 8];  // K 16-23 / 24-31
      af[mt] = __builtin_shufflevector(lo, hi,
          0, 1, 2, 3, 4, 5, 6, 7, 8, 9, 10, 11, 12, 13, 14, 15);
    }
#pragma unroll
    for (int t = 0; t < 4; ++t) {
      const int bRow = t * 16 + l16;
      v8bf lo = *(const v8bf*)&lB[bRow * LDSS + hi16 * 8];
      v8bf hi = *(const v8bf*)&lB[bRow * LDSS + 16 + hi16 * 8];
      v16bf bf_ = __builtin_shufflevector(lo, hi,
          0, 1, 2, 3, 4, 5, 6, 7, 8, 9, 10, 11, 12, 13, 14, 15);
#pragma unroll
      for (int mt = 0; mt < 2; ++mt) {
        acc[mt][t] = __builtin_amdgcn_wmma_f32_16x16x32_bf16(
            false, af[mt], false, bf_, (short)0, acc[mt][t], false, false);
      }
    }
  };

  stage(0, 0);                        // prefetch first K-tile
  int buf = 0;
  for (int k0 = 0; k0 < K; k0 += BK) {
    wait_async0();                    // own async stores to LDS landed
    __syncthreads();                  // everyone's landed; prev compute done
    if (k0 + BK < K) stage(buf ^ 1, k0 + BK);
    compute(buf);
    buf ^= 1;
  }

  // ---- writeback: D layout => VGPR r: M = r + hi16*8, N = lane&15
#pragma unroll
  for (int mt = 0; mt < 2; ++mt) {
#pragma unroll
    for (int t = 0; t < 4; ++t) {
#pragma unroll
      for (int r = 0; r < 8; ++r) {
        const long row = mBlock + wave * 32 + mt * 16 + hi16 * 8 + r;
        const long col = nBlock + t * 16 + l16;
        float v = acc[mt][t][r] * alpha;
        if (ADD_BIAS) v += bias[col];
        if (OUT_BF16) {
          __bf16* Cb = (__bf16*)Cv + (long)bz * strideC;
          if (OUT_T) Cb[col * ldC + row] = (__bf16)v;
          else       Cb[row * ldC + col] = (__bf16)v;
        } else {
          float* Cb = (float*)Cv + (long)bz * strideC;
          Cb[row * ldC + col] = v;
        }
      }
    }
  }
}

// ---------------------------------------------------------------------------
// softmax over axis=1 (the QUERY axis): one thread per (batch, key-column).
// Adjacent threads touch adjacent columns of the same row -> coalesced.
// ---------------------------------------------------------------------------
__global__ __launch_bounds__(256)
void softmax_axis1_kernel(const float* __restrict__ scores,
                          __bf16* __restrict__ attn, int S) {
  const long idx = (long)blockIdx.x * blockDim.x + threadIdx.x;
  const int b   = (int)(idx / S);
  const int col = (int)(idx % S);
  const float* p = scores + (long)b * S * S + col;
  __bf16*      o = attn   + (long)b * S * S + col;

  float m = -3.4e38f;
  for (int q = 0; q < S; ++q) m = fmaxf(m, p[(long)q * S]);
  float s = 0.f;
  for (int q = 0; q < S; ++q) s += __expf(p[(long)q * S] - m);
  const float inv = 1.f / s;
  for (int q = 0; q < S; ++q)
    o[(long)q * S] = (__bf16)(__expf(p[(long)q * S] - m) * inv);
}

// ---------------------------------------------------------------------------
extern "C" void kernel_launch(void* const* d_in, const int* in_sizes, int n_in,
                              void* d_out, int out_size, void* d_ws, size_t ws_size,
                              hipStream_t stream) {
  const int B = 4, S = 2048, D = 1024, H = 1024;
  const long BS = (long)B * S;
  const float SCALE = 0.125f;   // 64^-0.5

  const float* query = (const float*)d_in[0];
  const float* key_  = (const float*)d_in[1];
  const float* value = (const float*)d_in[2];
  const float* Wq = (const float*)d_in[3];
  const float* bq = (const float*)d_in[4];
  const float* Wk = (const float*)d_in[5];
  const float* bk = (const float*)d_in[6];
  const float* Wv = (const float*)d_in[7];
  const float* bv = (const float*)d_in[8];
  const float* Wo = (const float*)d_in[9];
  const float* bo = (const float*)d_in[10];
  float* out = (float*)d_out;

  size_t off = 0;
  auto take = [&](size_t bytes) -> void* {
    off = (off + 255) & ~(size_t)255;
    void* p = (char*)d_ws + off;
    off += bytes;
    return p;
  };

  __bf16* xq  = (__bf16*)take(BS * D * sizeof(__bf16));
  __bf16* xk  = (__bf16*)take(BS * D * sizeof(__bf16));
  __bf16* xv  = (__bf16*)take(BS * D * sizeof(__bf16));
  __bf16* wq  = (__bf16*)take((long)H * D * sizeof(__bf16));
  __bf16* wk  = (__bf16*)take((long)H * D * sizeof(__bf16));
  __bf16* wv  = (__bf16*)take((long)H * D * sizeof(__bf16));
  __bf16* wo  = (__bf16*)take((long)D * H * sizeof(__bf16));
  __bf16* qb  = (__bf16*)take(BS * H * sizeof(__bf16));
  __bf16* kb  = (__bf16*)take(BS * H * sizeof(__bf16));
  __bf16* vbT = (__bf16*)take((long)H * BS * sizeof(__bf16));  // [H, B*S]
  float*  sc  = (float*) take((long)B * S * S * sizeof(float));
  __bf16* at  = (__bf16*)take((long)B * S * S * sizeof(__bf16));
  __bf16* cx  = (__bf16*)take(BS * H * sizeof(__bf16));

  // ---- fp32 -> bf16 conversions ----
  cvt_f32_to_bf16_kernel<<<1024, 128, 0, stream>>>(query, xq, BS * D);
  cvt_f32_to_bf16_kernel<<<1024, 128, 0, stream>>>(key_,  xk, BS * D);
  cvt_f32_to_bf16_kernel<<<1024, 128, 0, stream>>>(value, xv, BS * D);
  cvt_f32_to_bf16_kernel<<<256,  128, 0, stream>>>(Wq, wq, (long)H * D);
  cvt_f32_to_bf16_kernel<<<256,  128, 0, stream>>>(Wk, wk, (long)H * D);
  cvt_f32_to_bf16_kernel<<<256,  128, 0, stream>>>(Wv, wv, (long)H * D);
  cvt_f32_to_bf16_kernel<<<256,  128, 0, stream>>>(Wo, wo, (long)D * H);

  // ---- q/k projections: [BS,D] x [H,D]^T + bias -> bf16 [BS,H] ----
  {
    dim3 grid(H / BN, (unsigned)(BS / BM), 1);
    gemm_nt_wmma_bf16_kernel<true, true, false><<<grid, 128, 0, stream>>>(
        xq, D, 0, wq, D, 0, qb, H, 0, bq, D, 1.0f);
    gemm_nt_wmma_bf16_kernel<true, true, false><<<grid, 128, 0, stream>>>(
        xk, D, 0, wk, D, 0, kb, H, 0, bk, D, 1.0f);
    // v projection with transposed writeback -> vbT [H, BS]
    gemm_nt_wmma_bf16_kernel<true, true, true><<<grid, 128, 0, stream>>>(
        xv, D, 0, wv, D, 0, vbT, BS, 0, bv, D, 1.0f);
  }

  // ---- scores = q k^T * SCALE : batched [S,H] x [S,H]^T -> fp32 [S,S] ----
  {
    dim3 grid(S / BN, S / BM, B);
    gemm_nt_wmma_bf16_kernel<false, false, false><<<grid, 128, 0, stream>>>(
        qb, H, (long)S * H, kb, H, (long)S * H, sc, S, (long)S * S, nullptr,
        H, SCALE);
  }

  // ---- softmax over query axis -> bf16 attn ----
  softmax_axis1_kernel<<<(unsigned)((long)B * S / 256), 256, 0, stream>>>(sc, at, S);

  // ---- ctx = attn @ V : NT against vbT. B row n lives at vbT + n*BS + b*S ----
  {
    dim3 grid(H / BN, S / BM, B);
    gemm_nt_wmma_bf16_kernel<true, false, false><<<grid, 128, 0, stream>>>(
        at, S, (long)S * S, vbT, BS, (long)S, cx, H, (long)S * H, nullptr,
        S, 1.0f);
  }

  // ---- out = ctx @ Wo^T + bo : [BS,H] x [D,H]^T -> fp32 [BS,D] ----
  {
    dim3 grid(D / BN, (unsigned)(BS / BM), 1);
    gemm_nt_wmma_bf16_kernel<false, true, false><<<grid, 128, 0, stream>>>(
        cx, H, 0, wo, H, 0, out, D, 0, bo, H, 1.0f);
  }
}
